// LightGlueAttention_50672024158505
// MI455X (gfx1250) — matrix operand order repaired and checked by reading, not compiled, not run on hardware
//
#include <hip/hip_runtime.h>
#include <hip/hip_bf16.h>

// CDNA5 / gfx1250 attention (B,H,N,D = 4,4,2048,64), fp32 in/out.
// d_out = [output (B*H*N*D)] ++ [attention (B*H*N*N)].
// Pre-convert K (row-major bf16) and V (transposed bf16) into d_ws; per
// (head, 16-query tile) workgroup: QK^T via v_wmma_f32_16x16x32_bf16 into an
// f32 LDS panel, vectorized masked softmax in LDS (attention streamed out
// non-temporally, P kept bf16 in LDS), then P.V via WMMA with b128 loads only.
// All hot-loop addresses are single-base + immediate-offset.

typedef __attribute__((ext_vector_type(16))) __bf16        v16bf;
typedef __attribute__((ext_vector_type(8)))  float         v8f;
typedef __attribute__((ext_vector_type(4)))  unsigned int  u32x4;
typedef __attribute__((ext_vector_type(2)))  unsigned int  u32x2;
typedef __attribute__((ext_vector_type(4)))  float         f32x4;
typedef __attribute__((ext_vector_type(4)))  int           i32x4;

#define BATCH  4
#define HEADS  4
#define SEQ    2048
#define DIM    64
#define QTILE  16
#define NWAVES 8
#define SROW   2052                  // f32 score row stride (+4 pad; 8208B, 16B-aligned)
#define PROW   1028                  // u32 bf16-pair row stride (+4 pad; keeps 16B align)

union Frag16 {
  unsigned int u[8];
  u32x4        q[2];
  v16bf        v;
};

// one-VALU f32->bf16 pair pack (truncate): result = (hi>>16)<<16 | (lo>>16)
__device__ __forceinline__ unsigned int pack_bf16(float lo, float hi) {
  return __builtin_amdgcn_perm(__float_as_uint(hi), __float_as_uint(lo), 0x07060302u);
}

// ---------------- prep: K f32 -> bf16 row-major -------------------------------
__global__ void __launch_bounds__(256)
cvt_k_kernel(const float* __restrict__ k, unsigned int* __restrict__ kb) {
  size_t i = (size_t)blockIdx.x * 256 + threadIdx.x;      // one bf16 pair per thread
  kb[i] = pack_bf16(k[2 * i], k[2 * i + 1]);
}

// ---------------- prep: V f32 -> bf16, transposed to [bh][d][k] ---------------
__global__ void __launch_bounds__(256)
cvt_vt_kernel(const float* __restrict__ v, unsigned int* __restrict__ vt) {
  __shared__ float tile[DIM][65];                          // 64x64 tile, padded
  const int bh = blockIdx.x >> 5;                          // 32 n-tiles of 64
  const int nt = blockIdx.x & 31;
  const int t  = threadIdx.x;
  const float* vb = v + ((size_t)bh * SEQ + nt * 64) * DIM;
  #pragma unroll
  for (int r = 0; r < 16; ++r) {                           // coalesced reads
    int idx = t + 256 * r;                                 // 0..4095
    int n = idx >> 6, d = idx & 63;
    tile[d][n] = vb[n * DIM + d];
  }
  __syncthreads();
  unsigned int* ot = vt + (size_t)bh * DIM * (SEQ / 2);    // row stride 1024 u32
  #pragma unroll
  for (int r = 0; r < 8; ++r) {                            // coalesced packed writes
    int o = t + 256 * r;                                   // 0..2047
    int d = o >> 5, n2 = o & 31;
    ot[(size_t)d * (SEQ / 2) + nt * 32 + n2] = pack_bf16(tile[d][2 * n2], tile[d][2 * n2 + 1]);
  }
}

// ---------------- main attention kernel ---------------------------------------
__global__ void __launch_bounds__(256)
lightglue_attn_kernel(const float* __restrict__ q,
                      const unsigned int* __restrict__ kbf,  // [bh][row][32] u32 (bf16x2)
                      const unsigned int* __restrict__ vt,   // [bh][d][1024] u32 (bf16x2)
                      const int*   __restrict__ mask,
                      float* __restrict__ out,               // B*H*N*D
                      float* __restrict__ attn) {            // B*H*N*N
  extern __shared__ float lds[];
  float*        S   = lds;                                   // 16 x SROW f32 scores
  unsigned int* PB  = (unsigned int*)(lds + QTILE * SROW);   // 16 x PROW u32 bf16 probs
  float*        RED = lds + QTILE * SROW + QTILE * PROW;     // 8*8*32 f32 partial O

  const int tid  = threadIdx.x;
  const int lane = tid & 31;
  const int wave = tid >> 5;
  const int half = lane >> 4;
  const int l16  = lane & 15;

  const int blk = blockIdx.x;                // bh*128 + qtile
  const int qt  = blk & 127;
  const int bh  = blk >> 7;
  const int b   = bh >> 2;                   // H == 4
  const int q0  = qt * QTILE;

  const float* Qb = q + (size_t)bh * SEQ * DIM;
  const int*   mb = mask + b * SEQ;

  // ---- persistent Q A-fragments (scale 1/sqrt(64) folded in) ----
  const float scale = 0.125f;
  Frag16 aq[2];
  {
    const float* Qrow = Qb + (size_t)(q0 + l16) * DIM;
    #pragma unroll
    for (int c = 0; c < 2; ++c) {
      #pragma unroll
      for (int i = 0; i < 8; ++i) {
        int kk = c * 32 + ((i < 4) ? (half * 8 + 2 * i) : (16 + half * 8 + 2 * (i - 4)));
        aq[c].u[i] = pack_bf16(Qrow[kk] * scale, Qrow[kk + 1] * scale);
      }
    }
  }

  // ---- Phase 1: S = (Q*scale) . K^T ; each wave does 16 contiguous k-tiles ----
  {
    // one global base (folds wave, l16, half); tile/chunk offsets are immediates
    const unsigned int* kb0 =
        kbf + ((size_t)bh * SEQ + wave * 256 + l16) * 32 + half * 4;
    // one LDS base for the score stores; offsets r*SROW + t*16 are immediates
    float* sb0 = S + (8 * half) * SROW + wave * 256 + l16;
    #pragma unroll
    for (int t = 0; t < 16; ++t) {
      v8f acc = {0.f, 0.f, 0.f, 0.f, 0.f, 0.f, 0.f, 0.f};
      #pragma unroll
      for (int c = 0; c < 2; ++c) {
        Frag16 bf;
        bf.q[0] = *(const u32x4*)(kb0 + t * 512 + c * 16);      // K = c*32+half*8+0..7
        bf.q[1] = *(const u32x4*)(kb0 + t * 512 + c * 16 + 8);  // K = c*32+16+half*8+0..7
        acc = __builtin_amdgcn_wmma_f32_16x16x32_bf16(false, aq[c].v, false, bf.v,
                                                      (short)0, acc, false, false);
      }
      #pragma unroll
      for (int r = 0; r < 8; ++r)             // D: vgpr r -> M = r+8*half, N = l16
        sb0[r * SROW + t * 16] = acc[r];
    }
  }
  __syncthreads();

  // ---- Phase 2: masked softmax, fully vectorized (b128 LDS / global ops) ----
  {
    const int row = tid >> 4;                 // 16 threads per row
    const int sub = tid & 15;
    float* s0 = S + row * SROW + sub * 4;     // hoisted bases; c*64 offsets
    const int* mb0 = mb + sub * 4;
    const float NEG = -3.402823466e38f;       // jnp.finfo(f32).min semantics

    // Pass A: apply mask once (write back), track row max
    float mmax = NEG;
    #pragma unroll 4
    for (int c = 0; c < SEQ / 64; ++c) {
      f32x4 s4 = *(f32x4*)(s0 + c * 64);
      i32x4 m4 = *(const i32x4*)(mb0 + c * 64);
      s4.x = (m4.x != 0) ? s4.x : NEG;
      s4.y = (m4.y != 0) ? s4.y : NEG;
      s4.z = (m4.z != 0) ? s4.z : NEG;
      s4.w = (m4.w != 0) ? s4.w : NEG;
      mmax = fmaxf(mmax, fmaxf(fmaxf(s4.x, s4.y), fmaxf(s4.z, s4.w)));
      *(f32x4*)(s0 + c * 64) = s4;
    }
    #pragma unroll
    for (int off = 8; off > 0; off >>= 1)
      mmax = fmaxf(mmax, __shfl_xor(mmax, off, 16));

    // Pass B: e = exp(s - m) once per element (write back), accumulate sum
    float lsum = 0.f;
    #pragma unroll 4
    for (int c = 0; c < SEQ / 64; ++c) {
      f32x4 s4 = *(f32x4*)(s0 + c * 64);
      f32x4 e4;
      e4.x = __expf(s4.x - mmax);
      e4.y = __expf(s4.y - mmax);
      e4.z = __expf(s4.z - mmax);
      e4.w = __expf(s4.w - mmax);
      lsum += (e4.x + e4.y) + (e4.z + e4.w);
      *(f32x4*)(s0 + c * 64) = e4;
    }
    #pragma unroll
    for (int off = 8; off > 0; off >>= 1)
      lsum += __shfl_xor(lsum, off, 16);
    const float inv = 1.f / lsum;

    // Pass C: normalize, stream attention out (NT), pack bf16 P into LDS
    float* a0 = attn + ((size_t)bh * SEQ + (q0 + row)) * SEQ + sub * 4;
    unsigned int* p0 = PB + row * PROW + sub * 2;
    #pragma unroll 4
    for (int c = 0; c < SEQ / 64; ++c) {
      f32x4 e4 = *(f32x4*)(s0 + c * 64);
      f32x4 p4;
      p4.x = e4.x * inv;  p4.y = e4.y * inv;
      p4.z = e4.z * inv;  p4.w = e4.w * inv;
      __builtin_nontemporal_store(p4, (f32x4*)(a0 + c * 64));   // 268 MB stream
      u32x2 pk;
      pk.x = pack_bf16(p4.x, p4.y);
      pk.y = pack_bf16(p4.z, p4.w);
      *(u32x2*)(p0 + c * 32) = pk;
    }
  }
  __syncthreads();

  // ---- Phase 3: O = P . V ; 8 waves = 4 n-tiles x 2 K-halves, b128 loads only ----
  {
    const int nt    = wave & 3;
    const int khalf = wave >> 2;
    const int kb2   = khalf * (SEQ / 4);          // kbase/2 in u32 units
    // hoisted bases: chunk offsets c*16 are immediates
    const unsigned int* vb0 =
        vt + ((size_t)bh * DIM + nt * 16 + l16) * (SEQ / 2) + kb2 + half * 4;
    const unsigned int* pb0 = PB + l16 * PROW + kb2 + half * 4;
    v8f acc = {0.f, 0.f, 0.f, 0.f, 0.f, 0.f, 0.f, 0.f};
    #pragma unroll
    for (int c = 0; c < (SEQ / 2) / 32; ++c) {
      Frag16 af, bf;
      af.q[0] = *(const u32x4*)(pb0 + c * 16);      // K = koff + half*8  + 0..7
      af.q[1] = *(const u32x4*)(pb0 + c * 16 + 8);  // K = koff + 16 + half*8 + 0..7
      bf.q[0] = *(const u32x4*)(vb0 + c * 16);      // K = koff + half*8  + 0..7
      bf.q[1] = *(const u32x4*)(vb0 + c * 16 + 8);  // K = koff + 16 + half*8 + 0..7
      acc = __builtin_amdgcn_wmma_f32_16x16x32_bf16(false, af.v, false, bf.v,
                                                    (short)0, acc, false, false);
    }
    #pragma unroll
    for (int r = 0; r < 8; ++r)
      RED[(wave * 8 + r) * 32 + lane] = acc[r];
  }
  __syncthreads();

  // ---- reduce K-halves, write 16x64 output tile ----
  if (wave < 4) {
    float* Orow = out + ((size_t)bh * SEQ + q0) * DIM;
    #pragma unroll
    for (int r = 0; r < 8; ++r) {
      float vsum = RED[(wave * 8 + r) * 32 + lane] +
                   RED[((wave + 4) * 8 + r) * 32 + lane];
      int m = r + 8 * half;
      Orow[(size_t)m * DIM + wave * 16 + l16] = vsum;
    }
  }
}

extern "C" void kernel_launch(void* const* d_in, const int* in_sizes, int n_in,
                              void* d_out, int out_size, void* d_ws, size_t ws_size,
                              hipStream_t stream) {
  const float* q    = (const float*)d_in[0];
  const float* k    = (const float*)d_in[1];
  const float* v    = (const float*)d_in[2];
  const int*   mask = (const int*)d_in[3];

  float* out  = (float*)d_out;
  float* attn = out + (size_t)BATCH * HEADS * SEQ * DIM;

  // workspace: bf16 K (4 MB) + bf16 V^T (4 MB)
  const size_t HALF_ELEMS = (size_t)BATCH * HEADS * SEQ * DIM / 2;  // u32 count
  unsigned int* kbf = (unsigned int*)d_ws;
  unsigned int* vt  = kbf + HALF_ELEMS;

  cvt_k_kernel <<<(int)(HALF_ELEMS / 256), 256, 0, stream>>>(k, kbf);
  cvt_vt_kernel<<<BATCH * HEADS * (SEQ / 64), 256, 0, stream>>>(v, vt);

  const size_t shmem = (size_t)(QTILE * SROW + QTILE * PROW + NWAVES * 8 * 32) * sizeof(float);
  hipFuncSetAttribute((const void*)lightglue_attn_kernel,
                      hipFuncAttributeMaxDynamicSharedMemorySize, (int)shmem);

  const int blocks = BATCH * HEADS * (SEQ / QTILE);   // 2048
  lightglue_attn_kernel<<<blocks, 256, shmem, stream>>>(q, kbf, vt, mask, out, attn);
}
